// MHAttention_66786741452940
// MI455X (gfx1250) — matrix operand order repaired and checked by reading, not compiled, not run on hardware
//
#include <hip/hip_runtime.h>
#include <math.h>

// Problem constants (match reference)
#define BB 2
#define SS 2048
#define DD 1024
#define HH 16
#define DKK 64
#define HD  (HH * DKK)   // 1024
#define BS  (BB * SS)    // 4096
#define BIAS_STRIDE 4096 // padded 2*SS-1 = 4095

#define USE_ASYNC_STAGING 1  // CDNA5 GLOBAL_LOAD_ASYNC_TO_LDS_* path

typedef __bf16 bf16_t;
typedef bf16_t bf16x8  __attribute__((ext_vector_type(8)));
typedef bf16_t bf16x16 __attribute__((ext_vector_type(16)));
typedef float  f32x8   __attribute__((ext_vector_type(8)));

static __device__ __forceinline__ f32x8 wmma_bf16(bf16x16 a, bf16x16 b, f32x8 c) {
  // D = A(16x32 bf16) x B(32x16 bf16) + C(16x16 f32)
  return __builtin_amdgcn_wmma_f32_16x16x32_bf16(false, a, false, b, (short)0, c,
                                                 false, false);
}

static __device__ __forceinline__ bf16x16 join8(bf16x8 lo, bf16x8 hi) {
  return __builtin_shufflevector(lo, hi, 0, 1, 2, 3, 4, 5, 6, 7,
                                 8, 9, 10, 11, 12, 13, 14, 15);
}

// A fragment (16x32 bf16). Lane holds row M = lane&15.
// lanes 0-15:  K[0..15]
// lanes 16-31: K[16..31]
static __device__ __forceinline__ bf16x16 load_afrag(const bf16_t* row, int hl) {
  bf16x8 lo = *(const bf16x8*)(row + 16 * hl);
  bf16x8 hi = *(const bf16x8*)(row + 16 * hl + 8);
  return join8(lo, hi);
}

// B fragment (32x16 bf16). Lane holds col N = lane&15;
// lanes 0-15 hold K=[0..15], lanes 16-31 hold K=[16..31].
static __device__ __forceinline__ bf16x16 load_bfrag(const bf16_t* p) {
  bf16x8 lo = *(const bf16x8*)(p);
  bf16x8 hi = *(const bf16x8*)(p + 8);
  return join8(lo, hi);
}

#if USE_ASYNC_STAGING
// CDNA5 async copy global -> LDS (16B per lane), tracked by ASYNCcnt.
// Generic LDS pointer's low 32 bits are the LDS byte address (ISA 10.2).
static __device__ __forceinline__ void async_copy_b128(void* lds,
                                                       const void* gaddr) {
  unsigned loff = (unsigned)(size_t)lds;
  unsigned long long ga = (unsigned long long)(size_t)gaddr;
  asm volatile("global_load_async_to_lds_b128 %0, %1, off" ::"v"(loff), "v"(ga)
               : "memory");
}
static __device__ __forceinline__ void wait_async0() {
  asm volatile("s_wait_asynccnt 0x0" ::: "memory");
}
#endif

// ---------------------------------------------------------------- conversions
__global__ void cvt_f32_bf16_x4(const float* __restrict__ src,
                                bf16_t* __restrict__ dst, int n4) {
  int i = blockIdx.x * blockDim.x + threadIdx.x;
  if (i >= n4) return;
  float4 v = ((const float4*)src)[i];
  dst[4 * i + 0] = (bf16_t)v.x;
  dst[4 * i + 1] = (bf16_t)v.y;
  dst[4 * i + 2] = (bf16_t)v.z;
  dst[4 * i + 3] = (bf16_t)v.w;
}

// f32 [K][N] -> bf16 [N][K] (32x32 LDS-tiled transpose, coalesced both sides)
__global__ __launch_bounds__(256) void cvt_transpose_bf16(
    const float* __restrict__ src, bf16_t* __restrict__ dst, int K, int N) {
  __shared__ float tile[32][33];
  int k0 = blockIdx.x * 32, n0 = blockIdx.y * 32;
  int tx = threadIdx.x & 31, ty = threadIdx.x >> 5;  // ty in 0..7
#pragma unroll
  for (int i = ty; i < 32; i += 8)
    tile[i][tx] = src[(size_t)(k0 + i) * N + n0 + tx];
  __syncthreads();
#pragma unroll
  for (int i = ty; i < 32; i += 8)
    dst[(size_t)(n0 + i) * K + k0 + tx] = (bf16_t)tile[tx][i];
}

// --------------------------------------------------------- T5 bias LUT [H][4096]
__global__ void bias_table_kernel(const float* __restrict__ rel_bias,
                                  float* __restrict__ table) {
  int idx = blockIdx.x * blockDim.x + threadIdx.x;
  if (idx >= HH * BIAS_STRIDE) return;
  int h = idx >> 12, d = idx & (BIAS_STRIDE - 1);
  if (d >= 2 * SS - 1) { table[idx] = 0.f; return; }
  int delta = d - (SS - 1);       // k - q
  int n = -delta;                 // q - k
  int ret = (n < 0) ? 16 : 0;     // num_buckets//2 = 16
  int an = (n < 0) ? -n : n;
  int v;
  if (an < 8) {                   // max_exact = 8
    v = an;
  } else {
    v = 8 + (int)(__logf((float)an * 0.125f) / __logf(16.0f) * 8.0f);
    if (v > 15) v = 15;
  }
  table[idx] = rel_bias[(ret + v) * HH + h];  // rel_bias[bucket][h]
}

// ------------------------------------------------------------- WMMA GEMM
// C = A(MxK bf16, row-major) * Wt(NxK bf16, i.e. W pre-transposed)
// Double-buffered async staging: GLOBAL_LOAD_ASYNC_TO_LDS_B128 fills buf^1
// while WMMAs consume buf. One s_wait_asynccnt + one barrier per K-step.
// MODE 0: write bf16 to [b,h,s,dk]   (q / k buffers)
// MODE 1: write bf16 to [b,h,dk,s]   (v transposed)
// MODE 2: write f32 row-major MxN    (final output)
template <int MODE>
__global__ __launch_bounds__(128) void gemm_bf16_wmma(
    const bf16_t* __restrict__ A, const bf16_t* __restrict__ Wt,
    void* __restrict__ dst, int M, int N, int K) {
  __shared__ __align__(16) bf16_t lds_a[2][64][40];  // pad: conflict-free b128
  __shared__ __align__(16) bf16_t lds_b[2][64][40];  // Wt rows: [n][k]

  const int tid = threadIdx.x;
  const int wv = tid >> 5;
  const int lane = tid & 31;
  const int hl = lane >> 4;
  const int ln = lane & 15;
  const int m0 = blockIdx.x * 64;
  const int n0 = blockIdx.y * 64;

  const int srow = tid >> 1, skh = (tid & 1) * 16;  // staging assignment
  const bf16_t* ap = A + (size_t)(m0 + srow) * K + skh;
  const bf16_t* bp = Wt + (size_t)(n0 + srow) * K + skh;

#if USE_ASYNC_STAGING
  auto issue = [&](int buf, int k0) {
    async_copy_b128(&lds_a[buf][srow][skh], ap + k0);
    async_copy_b128(&lds_a[buf][srow][skh + 8], ap + k0 + 8);
    async_copy_b128(&lds_b[buf][srow][skh], bp + k0);
    async_copy_b128(&lds_b[buf][srow][skh + 8], bp + k0 + 8);
  };
#else
  auto issue = [&](int buf, int k0) {
    *(bf16x8*)(&lds_a[buf][srow][skh])     = *(const bf16x8*)(ap + k0);
    *(bf16x8*)(&lds_a[buf][srow][skh + 8]) = *(const bf16x8*)(ap + k0 + 8);
    *(bf16x8*)(&lds_b[buf][srow][skh])     = *(const bf16x8*)(bp + k0);
    *(bf16x8*)(&lds_b[buf][srow][skh + 8]) = *(const bf16x8*)(bp + k0 + 8);
  };
#endif

  f32x8 acc[4] = {};
  issue(0, 0);
  int cur = 0;
  for (int k0 = 0; k0 < K; k0 += 32) {
#if USE_ASYNC_STAGING
    wait_async0();          // my copies for lds_*[cur] have landed
#endif
    __syncthreads();        // all waves' copies landed; buf^1 no longer read
    if (k0 + 32 < K) issue(cur ^ 1, k0 + 32);  // overlap fill with compute

    bf16x16 aF = load_afrag(&lds_a[cur][wv * 16 + ln][0], hl);
#pragma unroll
    for (int c = 0; c < 4; ++c) {
      bf16x16 bF = load_bfrag(&lds_b[cur][c * 16 + ln][16 * hl]);
      acc[c] = wmma_bf16(aF, bF, acc[c]);
    }
    cur ^= 1;
  }

  // epilogue: C layout => lane holds col n=ln, VGPR r holds row 8*hl + r
#pragma unroll
  for (int c = 0; c < 4; ++c) {
#pragma unroll
    for (int r = 0; r < 8; ++r) {
      int gm = m0 + wv * 16 + hl * 8 + r;
      int gn = n0 + c * 16 + ln;
      float v = acc[c][r];
      if (MODE == 2) {
        ((float*)dst)[(size_t)gm * N + gn] = v;
      } else {
        int b = gm >> 11, s = gm & (SS - 1);
        int h = gn >> 6, dk = gn & (DKK - 1);
        bf16_t bv = (bf16_t)v;
        if (MODE == 0)
          ((bf16_t*)dst)[(((size_t)(b * HH + h) * SS + s) * DKK) + dk] = bv;
        else
          ((bf16_t*)dst)[(((size_t)(b * HH + h) * DKK + dk) * SS) + s] = bv;
      }
    }
  }
}

// ------------------------------------------------- flash attention (1 wave / 16q)
// 128-key blocks: 16 score WMMAs, one batched max reduction, exp, then
// 16 PV WMMAs + 4 rowsum WMMAs (P x ones). Softmax sums come from the matrix
// pipe, not lane shuffles.
__global__ __launch_bounds__(256) void flash_attn_kernel(
    const bf16_t* __restrict__ qb,   // [b,h,s,dk] bf16
    const bf16_t* __restrict__ kb,   // [b,h,s,dk] bf16
    const bf16_t* __restrict__ vtb,  // [b,h,dk,s] bf16
    const float* __restrict__ btab,  // [h][4096] bias(delta = k-q+2047)
    bf16_t* __restrict__ ctx) {      // [b,s,h*dk] bf16
  __shared__ __align__(16) bf16_t lds_p[8][16][136];  // per-wave P tile 16x128

  const int tid = threadIdx.x;
  const int wv = tid >> 5;
  const int lane = tid & 31;
  const int hl = lane >> 4;
  const int ln = lane & 15;

  const int gw = blockIdx.x * 8 + wv;   // 4096 waves total
  const int qt = gw & 127;              // S/16
  const int bh = gw >> 7;               // b*H + h
  const int h = bh & (HH - 1);
  const int q0 = qt * 16;

  const bf16_t* qbase = qb + (size_t)bh * SS * DKK;
  const bf16_t* kbase = kb + (size_t)bh * SS * DKK;
  const bf16_t* vbase = vtb + (size_t)bh * DKK * SS;
  const float* bias = btab + (size_t)h * BIAS_STRIDE + (SS - 1);

  // resident Q fragments (dk 0..31 and 32..63); lane's row = q0 + ln
  const bf16_t* qrow = qbase + (size_t)(q0 + ln) * DKK;
  const bf16x16 aq0 = load_afrag(qrow, hl);
  const bf16x16 aq1 = load_afrag(qrow + 32, hl);

  bf16x16 onesb;  // B = ones(32x16): WMMA(P, ones) -> per-row sum in C layout
#pragma unroll
  for (int i = 0; i < 16; ++i) onesb[i] = (bf16_t)1.0f;

  f32x8 o[4] = {};
  f32x8 lacc = {};   // running softmax denominators (row r == o's row 8*hl+r)
  float rm[8];
#pragma unroll
  for (int r = 0; r < 8; ++r) rm[r] = -__builtin_inff();
  const int qrow0 = q0 + hl * 8;  // global q row for VGPR slot r

  for (int kt = 0; kt < SS; kt += 128) {
    // ---- scores: eight 16x16 tiles over 128 keys (K=dk accumulated twice)
    f32x8 s[8];
#pragma unroll
    for (int j = 0; j < 8; ++j) {
      const bf16_t* kr = kbase + (size_t)(kt + j * 16 + ln) * DKK + 16 * hl;
      f32x8 t = {};
      t = wmma_bf16(aq0, load_bfrag(kr), t);
      t = wmma_bf16(aq1, load_bfrag(kr + 32), t);
      s[j] = t;
    }
    if (kt + 128 < SS) {  // global_prefetch_b8 next tiles
      __builtin_prefetch(kbase + (size_t)(kt + 128 + ln) * DKK, 0, 1);
      __builtin_prefetch(vbase + (size_t)ln * SS + kt + 128, 0, 1);
    }

    // ---- bias + batched online-softmax max (one shuffle chain per row)
    float scr[8];
#pragma unroll
    for (int r = 0; r < 8; ++r) {
      const int q = qrow0 + r;
      float mx = -__builtin_inff();
#pragma unroll
      for (int j = 0; j < 8; ++j) {
        float x = s[j][r] + bias[(kt + j * 16 + ln) - q];
        s[j][r] = x;
        mx = fmaxf(mx, x);
      }
      mx = fmaxf(mx, __shfl_xor(mx, 1, 32));
      mx = fmaxf(mx, __shfl_xor(mx, 2, 32));
      mx = fmaxf(mx, __shfl_xor(mx, 4, 32));
      mx = fmaxf(mx, __shfl_xor(mx, 8, 32));
      float mnew = fmaxf(rm[r], mx);
      scr[r] = __expf(rm[r] - mnew);
      rm[r] = mnew;
#pragma unroll
      for (int j = 0; j < 8; ++j) s[j][r] = __expf(s[j][r] - mnew);
    }
#pragma unroll
    for (int r = 0; r < 8; ++r) {
      lacc[r] *= scr[r];
#pragma unroll
      for (int c = 0; c < 4; ++c) o[c][r] *= scr[r];
    }

    // ---- transpose P (C layout -> A layout) through this wave's LDS tile
#pragma unroll
    for (int j = 0; j < 8; ++j)
#pragma unroll
      for (int r = 0; r < 8; ++r)
        lds_p[wv][hl * 8 + r][j * 16 + ln] = (bf16_t)s[j][r];

    // ---- O += P x V and Lacc += P x ones (rowsum from the matrix pipe)
#pragma unroll
    for (int kk = 0; kk < 4; ++kk) {
      bf16x16 aP = load_afrag(&lds_p[wv][ln][kk * 32], hl);
      lacc = wmma_bf16(aP, onesb, lacc);
#pragma unroll
      for (int c = 0; c < 4; ++c) {
        const bf16_t* vr =
            vbase + (size_t)(c * 16 + ln) * SS + kt + kk * 32 + 16 * hl;
        o[c] = wmma_bf16(aP, load_bfrag(vr), o[c]);
      }
    }
  }

  // finalize: O / l, write ctx[b][s][h*64 + dk] as bf16 (coalesced over ln)
  const int b = bh >> 4;
#pragma unroll
  for (int r = 0; r < 8; ++r) {
    float inv = 1.0f / lacc[r];
    int s = qrow0 + r;
    bf16_t* dst = ctx + ((size_t)b * SS + s) * HD + h * DKK;
#pragma unroll
    for (int c = 0; c < 4; ++c) dst[c * 16 + ln] = (bf16_t)(o[c][r] * inv);
  }
}

// ------------------------------------------------------------------ launch
extern "C" void kernel_launch(void* const* d_in, const int* in_sizes, int n_in,
                              void* d_out, int out_size, void* d_ws,
                              size_t ws_size, hipStream_t stream) {
  (void)in_sizes; (void)n_in; (void)out_size; (void)ws_size;
  const float* x = (const float*)d_in[0];
  const float* wq = (const float*)d_in[1];
  const float* wk = (const float*)d_in[2];
  const float* wv = (const float*)d_in[3];
  const float* wo = (const float*)d_in[4];
  const float* rel_bias = (const float*)d_in[5];

  char* ws = (char*)d_ws;
  size_t off = 0;
  auto alloc = [&](size_t bytes) { char* p = ws + off; off += bytes; return p; };
  bf16_t* xb   = (bf16_t*)alloc((size_t)BS * DD * 2);
  bf16_t* wqt  = (bf16_t*)alloc((size_t)DD * HD * 2);  // [N][K] bf16
  bf16_t* wkt  = (bf16_t*)alloc((size_t)DD * HD * 2);
  bf16_t* wvt  = (bf16_t*)alloc((size_t)DD * HD * 2);
  bf16_t* wot  = (bf16_t*)alloc((size_t)HD * DD * 2);
  bf16_t* qbuf  = (bf16_t*)alloc((size_t)BB * HH * SS * DKK * 2);
  bf16_t* kbuf  = (bf16_t*)alloc((size_t)BB * HH * SS * DKK * 2);
  bf16_t* vtbuf = (bf16_t*)alloc((size_t)BB * HH * DKK * SS * 2);
  bf16_t* ctxb  = (bf16_t*)alloc((size_t)BS * HD * 2);
  float* btab   = (float*)alloc((size_t)HH * BIAS_STRIDE * 4);

  // x: f32 -> bf16; weights: f32 [K][N] -> bf16 [N][K]
  {
    int n4 = BS * DD / 4;
    cvt_f32_bf16_x4<<<(n4 + 255) / 256, 256, 0, stream>>>(x, xb, n4);
    dim3 tg(DD / 32, HD / 32);
    cvt_transpose_bf16<<<tg, 256, 0, stream>>>(wq, wqt, DD, HD);
    cvt_transpose_bf16<<<tg, 256, 0, stream>>>(wk, wkt, DD, HD);
    cvt_transpose_bf16<<<tg, 256, 0, stream>>>(wv, wvt, DD, HD);
    dim3 tg2(HD / 32, DD / 32);
    cvt_transpose_bf16<<<tg2, 256, 0, stream>>>(wo, wot, HD, DD);
  }
  // bias LUT
  bias_table_kernel<<<(HH * BIAS_STRIDE + 255) / 256, 256, 0, stream>>>(rel_bias,
                                                                        btab);
  // projections (M=4096, N=1024, K=1024), 64x64 block tiles
  dim3 g(BS / 64, HD / 64);
  gemm_bf16_wmma<0><<<g, 128, 0, stream>>>(xb, wqt, qbuf, BS, HD, DD);
  gemm_bf16_wmma<0><<<g, 128, 0, stream>>>(xb, wkt, kbuf, BS, HD, DD);
  gemm_bf16_wmma<1><<<g, 128, 0, stream>>>(xb, wvt, vtbuf, BS, HD, DD);

  // flash attention: 4096 waves = 512 blocks x 8 waves
  flash_attn_kernel<<<(BB * HH * (SS / 16)) / 8, 256, 0, stream>>>(
      qbuf, kbuf, vtbuf, btab, ctxb);

  // output projection -> f32
  dim3 g2(BS / 64, DD / 64);
  gemm_bf16_wmma<2><<<g2, 128, 0, stream>>>(ctxb, wot, d_out, BS, DD, HD);
}
